// ZipNerfField_89404039234224
// MI455X (gfx1250) — matrix operand order repaired and checked by reading, not compiled, not run on hardware
//
#include <hip/hip_runtime.h>
#include <hip/hip_fp16.h>

typedef __attribute__((ext_vector_type(16))) _Float16 v16h;
typedef __attribute__((ext_vector_type(8)))  _Float16 h8;
typedef __attribute__((ext_vector_type(8)))  float    v8f;

#define R_RAYS 4096
#define S_SAMP 32
#define NPTS   (R_RAYS * S_SAMP)     // 131072 points
#define NGAUSS 6
#define NLVL   10
#define TSIZE  (1u << 21)            // hash table entries per level

// ---------------------------------------------------------------------------
// Phase 1: contraction + erf downweight + trilinear hash-grid gather.
// One thread per (point, level). Output: f16 feats [NPTS][64] (40 valid + pad).
// ---------------------------------------------------------------------------
__global__ void encode_kernel(const float* __restrict__ means,
                              const float* __restrict__ stds,
                              const float* __restrict__ table,
                              _Float16* __restrict__ feats)
{
  int idx = blockIdx.x * blockDim.x + threadIdx.x;
  if (idx >= NPTS * NLVL) return;
  int p = idx / NLVL;
  int l = idx - p * NLVL;
  float res = 16.f * (float)(1 << l);

  float acc0 = 0.f, acc1 = 0.f, acc2 = 0.f, acc3 = 0.f;
  const float4* tab = (const float4*)table + (size_t)l * TSIZE;

  for (int s = 0; s < NGAUSS; ++s) {
    const float* mp = means + ((size_t)p * NGAUSS + s) * 3;
    float x = mp[0], y = mp[1], z = mp[2];
    float sd = stds[(size_t)p * NGAUSS + s];
    // mip-nerf-360 contraction + |det J|^(1/3) std scaling
    float r2 = fmaxf(x * x + y * y + z * z, 1e-8f);
    float r  = sqrtf(r2);
    float f  = (2.f * r - 1.f) / r2;
    bool inside = (r2 <= 1.f);
    float cx = inside ? x : f * x;
    float cy = inside ? y : f * y;
    float cz = inside ? z : f * z;
    float det = (f * f) / r2;
    float scale = inside ? 1.f : cbrtf(det);
    float sc = sd * scale * 0.5f;                 // bound=2 rescale
    float x0 = (cx * 0.5f + 1.f) * 0.5f;          // -> [0,1]
    float y0 = (cy * 0.5f + 1.f) * 0.5f;
    float z0 = (cz * 0.5f + 1.f) * 0.5f;
    float aw = erff(rsqrtf(8.f * sc * sc * res * res));

    float px = x0 * res, py = y0 * res, pz = z0 * res;
    float fx = floorf(px), fy = floorf(py), fz = floorf(pz);
    float wx = px - fx, wy = py - fy, wz = pz - fz;
    unsigned bx = (unsigned)(int)fx, by = (unsigned)(int)fy, bz = (unsigned)(int)fz;
    #pragma unroll
    for (int c = 0; c < 8; ++c) {
      unsigned ox = c & 1u, oy = (c >> 1) & 1u, oz = (c >> 2) & 1u;
      unsigned h = (bx + ox) ^ ((by + oy) * 2654435761u) ^ ((bz + oz) * 805459861u);
      unsigned id = h & (TSIZE - 1u);
      float4 tv = tab[id];
      float w = (ox ? wx : 1.f - wx) * (oy ? wy : 1.f - wy) * (oz ? wz : 1.f - wz) * aw;
      acc0 += w * tv.x; acc1 += w * tv.y; acc2 += w * tv.z; acc3 += w * tv.w;
    }
  }
  const float inv = 1.f / (float)NGAUSS;
  _Float16* o = feats + (size_t)p * 64 + l * 4;
  o[0] = (_Float16)(acc0 * inv); o[1] = (_Float16)(acc1 * inv);
  o[2] = (_Float16)(acc2 * inv); o[3] = (_Float16)(acc3 * inv);
  if (l < 6) {                                    // zero K padding cols 40..63
    _Float16* zp = feats + (size_t)p * 64 + 40 + l * 4;
    zp[0] = (_Float16)0.f; zp[1] = (_Float16)0.f;
    zp[2] = (_Float16)0.f; zp[3] = (_Float16)0.f;
  }
}

// ---------------------------------------------------------------------------
// Phase 2a: view-direction positional encoding per ray -> f16 [R][32] (27+pad)
// ---------------------------------------------------------------------------
__global__ void direnc_kernel(const float* __restrict__ vd, _Float16* __restrict__ de)
{
  int r = blockIdx.x * blockDim.x + threadIdx.x;
  if (r >= R_RAYS) return;
  float v0 = vd[r * 3], v1 = vd[r * 3 + 1], v2 = vd[r * 3 + 2];
  float e[32];
  e[0] = v0; e[1] = v1; e[2] = v2;
  float v[3] = { v0, v1, v2 };
  #pragma unroll
  for (int d = 0; d < 4; ++d) {
    float s = (float)(1 << d);
    #pragma unroll
    for (int k = 0; k < 3; ++k) {
      float xb = v[k] * s;
      e[3  + d * 3 + k] = sinf(xb);
      e[15 + d * 3 + k] = cosf(xb);   // sin(x + pi/2)
    }
  }
  for (int k = 27; k < 32; ++k) e[k] = 0.f;
  _Float16* o = de + (size_t)r * 32;
  for (int k = 0; k < 32; ++k) o[k] = (_Float16)e[k];
}

// ---------------------------------------------------------------------------
// Phase 2b: repack f32 weights [K][N] into f16 WMMA B-fragment order.
// Fragment block (N-tile, K-chunk) = 512 halfs; lane l, half j:
//   N = ntile*16 + (l & 15),  K = kchunk*32 + (l>=16 ? 8 : 0) + (j & 7) + (j>=8 ? 16 : 0)
// ---------------------------------------------------------------------------
__global__ void repack_kernel(const float* __restrict__ src, _Float16* __restrict__ dst,
                              int Kpad, int Npad, int Ksrc, int Nsrc)
{
  int idx = blockIdx.x * blockDim.x + threadIdx.x;
  if (idx >= Kpad * Npad) return;
  int blk = idx >> 9;
  int within = idx & 511;
  int l = within >> 4;
  int j = within & 15;
  int nKc = Kpad >> 5;
  int nt = blk / nKc;
  int kc = blk - nt * nKc;
  int k = kc * 32 + ((l >> 4) << 3) + (j & 7) + ((j >> 3) << 4);
  int n = nt * 16 + (l & 15);
  float v = (k < Ksrc && n < Nsrc) ? src[(size_t)k * Nsrc + n] : 0.f;
  dst[idx] = (_Float16)v;
}

// ---------------------------------------------------------------------------
// WMMA helpers.  A fragment from row-major LDS per ISA 16-bit A layout:
//   lane L: M = L&15;  halfs 0..7  = K[kc*32 + koff .. +7]
//                      halfs 8..15 = K[kc*32 + 16 + koff .. +7],  koff=(L>=16)?8:0
// ---------------------------------------------------------------------------
__device__ inline v16h load_a_frag(const _Float16* row)
{
  h8 lo = *(const h8*)(row);
  h8 hi = *(const h8*)(row + 16);
  v16h a;
  #pragma unroll
  for (int i = 0; i < 8; ++i) { a[i] = lo[i]; a[i + 8] = hi[i]; }
  return a;
}

// NR row-tiles share each B fragment (halves weight/L2 traffic per point).
template <int NT, int NR>
__device__ inline void mma_layer(v8f acc[NR][NT], const _Float16* aBase, int aStride,
                                 int nK, const _Float16* wbase, int nKtotal,
                                 int tile0, int kBase, int m, int koff, int lane)
{
  for (int kc = 0; kc < nK; ++kc) {
    v16h a[NR];
    #pragma unroll
    for (int r = 0; r < NR; ++r)
      a[r] = load_a_frag(aBase + (r * 16 + m) * aStride + kc * 32 + koff);
    #pragma unroll
    for (int t = 0; t < NT; ++t) {
      v16h b = *(const v16h*)(wbase +
          ((size_t)(tile0 + t) * nKtotal + (kBase + kc)) * 512 + lane * 16);
      #pragma unroll
      for (int r = 0; r < NR; ++r)
        acc[r][t] = __builtin_amdgcn_wmma_f32_16x16x32_f16(
            false, a[r], false, b, (short)0, acc[r][t], false, false);
    }
  }
}

// ---------------------------------------------------------------------------
// Phase 3: fused MLP chain. 128 threads (4 waves) handle 32 points
// (2 row-tiles x 16).  Wave w owns N-columns [w*64, w*64+64).
// ---------------------------------------------------------------------------
__global__ __launch_bounds__(128)
void mlp_kernel(const _Float16* __restrict__ feats, const _Float16* __restrict__ de,
                const _Float16* __restrict__ wAll,
                const float* __restrict__ b1v, const float* __restrict__ b2v,
                const float* __restrict__ bd0v, const float* __restrict__ bd1v,
                const float* __restrict__ brgbv, float* __restrict__ out)
{
  __shared__ _Float16 actA[32 * 64];    // feats tile       (K=64)   4 KB
  __shared__ _Float16 h1[32 * 64];      // layer1 out       (K=64)   4 KB
  __shared__ _Float16 x2[32 * 288];     // x(256) ++ de(32) (K=288) 18 KB
  __shared__ _Float16 ybuf[32 * 256];   // dir layer0 out           16 KB
  __shared__ _Float16 y2buf[32 * 256];  // dir layer1 out           16 KB

  const int tid = threadIdx.x;
  const int lane = tid & 31;
  const int wid = tid >> 5;
  const int m = lane & 15;
  const int koff = (lane >> 4) * 8;
  const int mbase = (lane >> 4) * 8;    // C/D: VGPR j -> row mbase + j
  const int p0 = blockIdx.x * 32;
  const int ray = p0 >> 5;              // 32-point tile == exactly one ray

  const _Float16* wL1  = wAll;                  // 64x64
  const _Float16* wL2  = wL1 + 64 * 64;         // 64x256
  const _Float16* wD0  = wL2 + 64 * 256;        // 288x256
  const _Float16* wD1  = wD0 + 288 * 256;       // 544x256
  const _Float16* wRGB = wD1 + 544 * 256;       // 256x16

  // stage feats tile (32*64 halfs = 256 h8) and dir-enc columns
  {
    const h8* src = (const h8*)(feats + (size_t)p0 * 64);
    h8* dst = (h8*)actA;
    dst[tid]       = src[tid];
    dst[tid + 128] = src[tid + 128];
  }
  {
    int rr = tid >> 2;                  // 0..31 row
    int cc = (tid & 3) * 8;             // 0,8,16,24
    *(h8*)(x2 + rr * 288 + 256 + cc) = *(const h8*)(de + (size_t)ray * 32 + cc);
  }
  __syncthreads();

  // ---- layer 1: relu(feats @ W1 + b1)  [32,64]x[64,64], 1 N-tile per wave
  {
    v8f c[2][1];
    for (int r = 0; r < 2; ++r) for (int i = 0; i < 8; ++i) c[r][0][i] = 0.f;
    mma_layer<1, 2>(c, actA, 64, 2, wL1, 2, wid, 0, m, koff, lane);
    int n = wid * 16 + m;
    float bv = b1v[n];
    #pragma unroll
    for (int r = 0; r < 2; ++r)
      #pragma unroll
      for (int j = 0; j < 8; ++j)
        h1[(r * 16 + mbase + j) * 64 + n] = (_Float16)fmaxf(c[r][0][j] + bv, 0.f);
  }
  __syncthreads();

  // ---- layer 2: x = h1 @ W2 + b2 (raw); density = softplus(x[:,0] - 1)
  {
    v8f c[2][4];
    for (int r = 0; r < 2; ++r) for (int t = 0; t < 4; ++t)
      for (int i = 0; i < 8; ++i) c[r][t][i] = 0.f;
    mma_layer<4, 2>(c, h1, 64, 2, wL2, 2, wid * 4, 0, m, koff, lane);
    #pragma unroll
    for (int t = 0; t < 4; ++t) {
      int n = (wid * 4 + t) * 16 + m;
      float bv = b2v[n];
      #pragma unroll
      for (int r = 0; r < 2; ++r)
        #pragma unroll
        for (int j = 0; j < 8; ++j) {
          float v = c[r][t][j] + bv;
          x2[(r * 16 + mbase + j) * 288 + n] = (_Float16)v;
          if (n == 0) {
            float u = v - 1.f;
            float dens = (u > 20.f) ? u : log1pf(expf(u));
            out[(size_t)(p0 + r * 16 + mbase + j) * 4] = dens;
          }
        }
    }
  }
  __syncthreads();

  // ---- dir layer 0: y = relu(inp @ Wd0 + bd0)  K=288
  {
    v8f c[2][4];
    for (int r = 0; r < 2; ++r) for (int t = 0; t < 4; ++t)
      for (int i = 0; i < 8; ++i) c[r][t][i] = 0.f;
    mma_layer<4, 2>(c, x2, 288, 9, wD0, 9, wid * 4, 0, m, koff, lane);
    #pragma unroll
    for (int t = 0; t < 4; ++t) {
      int n = (wid * 4 + t) * 16 + m;
      float bv = bd0v[n];
      #pragma unroll
      for (int r = 0; r < 2; ++r)
        #pragma unroll
        for (int j = 0; j < 8; ++j)
          ybuf[(r * 16 + mbase + j) * 256 + n] = (_Float16)fmaxf(c[r][t][j] + bv, 0.f);
    }
  }
  __syncthreads();

  // ---- dir layer 1: y2 = relu([y, inp] @ Wd1 + bd1)  K = 256 + 288 = 544
  {
    v8f c[2][4];
    for (int r = 0; r < 2; ++r) for (int t = 0; t < 4; ++t)
      for (int i = 0; i < 8; ++i) c[r][t][i] = 0.f;
    mma_layer<4, 2>(c, ybuf, 256, 8, wD1, 17, wid * 4, 0, m, koff, lane);
    mma_layer<4, 2>(c, x2, 288, 9, wD1, 17, wid * 4, 8, m, koff, lane);
    #pragma unroll
    for (int t = 0; t < 4; ++t) {
      int n = (wid * 4 + t) * 16 + m;
      float bv = bd1v[n];
      #pragma unroll
      for (int r = 0; r < 2; ++r)
        #pragma unroll
        for (int j = 0; j < 8; ++j)
          y2buf[(r * 16 + mbase + j) * 256 + n] = (_Float16)fmaxf(c[r][t][j] + bv, 0.f);
    }
  }
  __syncthreads();

  // ---- rgb: sigmoid(y2 @ Wrgb + brgb) with padding, cols 0..2
  if (wid == 0) {
    v8f c[2][1];
    for (int r = 0; r < 2; ++r) for (int i = 0; i < 8; ++i) c[r][0][i] = 0.f;
    mma_layer<1, 2>(c, y2buf, 256, 8, wRGB, 8, 0, 0, m, koff, lane);
    if (m < 3) {
      float bv = brgbv[m];
      #pragma unroll
      for (int r = 0; r < 2; ++r)
        #pragma unroll
        for (int j = 0; j < 8; ++j) {
          float v = c[r][0][j] + bv;
          float sg = 1.f / (1.f + expf(-v));
          sg = sg * 1.002f - 0.001f;
          out[(size_t)(p0 + r * 16 + mbase + j) * 4 + 1 + m] = sg;
        }
    }
  }
}

// ---------------------------------------------------------------------------
extern "C" void kernel_launch(void* const* d_in, const int* in_sizes, int n_in,
                              void* d_out, int out_size, void* d_ws, size_t ws_size,
                              hipStream_t stream)
{
  (void)in_sizes; (void)n_in; (void)out_size; (void)ws_size;
  const float* means    = (const float*)d_in[0];
  const float* stds     = (const float*)d_in[1];
  const float* viewdirs = (const float*)d_in[2];
  const float* table    = (const float*)d_in[3];
  const float* W1   = (const float*)d_in[4];  const float* b1   = (const float*)d_in[5];
  const float* W2   = (const float*)d_in[6];  const float* b2   = (const float*)d_in[7];
  const float* Wd0  = (const float*)d_in[8];  const float* bd0  = (const float*)d_in[9];
  const float* Wd1  = (const float*)d_in[10]; const float* bd1  = (const float*)d_in[11];
  const float* Wrgb = (const float*)d_in[12]; const float* brgb = (const float*)d_in[13];
  float* out = (float*)d_out;

  _Float16* ws    = (_Float16*)d_ws;
  _Float16* feats = ws;                               // NPTS*64
  _Float16* de    = feats + (size_t)NPTS * 64;        // R_RAYS*32
  _Float16* wAll  = de + (size_t)R_RAYS * 32;
  _Float16* wL1   = wAll;
  _Float16* wL2   = wL1 + 64 * 64;
  _Float16* wD0   = wL2 + 64 * 256;
  _Float16* wD1   = wD0 + 288 * 256;
  _Float16* wRGB  = wD1 + 544 * 256;

  // weight repack into WMMA B-fragment order (f32 -> f16, with K/N padding)
  repack_kernel<<<(64 * 64   + 255) / 256, 256, 0, stream>>>(W1,   wL1,  64,  64,  40,  64);
  repack_kernel<<<(64 * 256  + 255) / 256, 256, 0, stream>>>(W2,   wL2,  64,  256, 64,  256);
  repack_kernel<<<(288 * 256 + 255) / 256, 256, 0, stream>>>(Wd0,  wD0,  288, 256, 283, 256);
  repack_kernel<<<(544 * 256 + 255) / 256, 256, 0, stream>>>(Wd1,  wD1,  544, 256, 539, 256);
  repack_kernel<<<(256 * 16  + 255) / 256, 256, 0, stream>>>(Wrgb, wRGB, 256, 16,  256, 3);

  direnc_kernel<<<(R_RAYS + 63) / 64, 64, 0, stream>>>(viewdirs, de);
  encode_kernel<<<(NPTS * NLVL + 255) / 256, 256, 0, stream>>>(means, stds, table, feats);
  mlp_kernel<<<NPTS / 32, 128, 0, stream>>>(feats, de, wAll, b1, b2, bd0, bd1, brgb, out);
}
